// SpikingLinearLayer_18777597018229
// MI455X (gfx1250) — matrix operand order repaired and checked by reading, not compiled, not run on hardware
//
#include <hip/hip_runtime.h>

typedef __attribute__((ext_vector_type(16))) _Float16 v16h;
typedef __attribute__((ext_vector_type(8)))  _Float16 v8h;
typedef __attribute__((ext_vector_type(4)))  _Float16 v4h;
typedef __attribute__((ext_vector_type(8)))  float    v8f;
typedef __attribute__((ext_vector_type(4)))  float    v4f;

namespace {
constexpr int   STEPS = 20;
constexpr int   BATCH = 512;
constexpr int   DIM   = 2048;          // input_dim == output_dim
constexpr float AM    = 0.95f;         // 1 - dt/tau_m
constexpr float DTM   = 0.05f;         // dt/tau_m
constexpr float AS    = 0.8f;          // 1 - dt/tau_s
constexpr float VTH   = 1.0f;
}

// ---------------- f32 -> f16 bulk convert (4 elems / thread) ----------------
__global__ void cvt_f32_to_f16(const float* __restrict__ src,
                               _Float16* __restrict__ dst, size_t n4) {
    size_t i = (size_t)blockIdx.x * blockDim.x + threadIdx.x;
    if (i >= n4) return;
    v4f v = ((const v4f*)src)[i];
    v4h h = { (_Float16)v[0], (_Float16)v[1], (_Float16)v[2], (_Float16)v[3] };
    ((v4h*)dst)[i] = h;
}

// ---------------- fragment loaders (wave32 WMMA; this code's contiguous k order, same in A and B) --------------
// A 16x32 f16: lane holds row = lane%16; 16 consecutive K values
// at k0 + (lane>=16 ? 16 : 0).
__device__ __forceinline__ v16h ld_a_h(const _Float16* p) {
    v8h r0 = *(const v8h*)(p);
    v8h r1 = *(const v8h*)(p + 8);
    v16h a;
#pragma unroll
    for (int i = 0; i < 8; ++i) { a[i] = r0[i]; a[8 + i] = r1[i]; }
    return a;
}
__device__ __forceinline__ v16h ld_a_f(const float* p) {
    v16h a;
#pragma unroll
    for (int i = 0; i < 8; ++i) { a[i] = (_Float16)p[i]; a[8 + i] = (_Float16)p[8 + i]; }
    return a;
}
// B 32x16 f16 (B = W^T, W row-major): lane holds column n = lane%16;
// 16 consecutive K values at k0 + (lane>=16 ? 16 : 0).
__device__ __forceinline__ v16h ld_b_h(const _Float16* p) {
    return *(const v16h*)p;                // 32B, aligned
}
__device__ __forceinline__ v16h ld_b_f(const float* p) {
    v16h b;
#pragma unroll
    for (int i = 0; i < 16; ++i) b[i] = (_Float16)p[i];
    return b;
}

// ============================================================================
// Kernel A: batched GEMM  C[t] = x[t] @ W^T   (t parallel across blockIdx.z)
// block = 256 threads (8 waves), wave grid 2(M) x 4(N).
// Wave tile: 32(M) x 64(N) = 8 WMMA subtiles -> 12 b128 loads / 8 WMMA per k.
// All 6 fragments of a k-chunk are materialized before the WMMA burst so the
// loadcnt waits stagger (0x3..0x0) instead of serializing on one B register.
// Block tile: 64(M) x 256(N). grid = (2048/256, 512/64, 20) = (8, 8, 20).
// ============================================================================
template <bool XF16>
__global__ __launch_bounds__(256)
void gemm_xt_wmma(const void* __restrict__ xptr,
                  const _Float16* __restrict__ Wh,
                  float* __restrict__ C) {
    const int tid    = threadIdx.x;
    const int lane   = tid & 31;
    const int ln     = lane & 15;
    const int laneHi = lane >> 4;
    const int waveId = tid >> 5;
    const int waveM  = waveId & 1;         // 2 waves along M
    const int waveN  = waveId >> 1;        // 4 waves along N
    const int t      = blockIdx.z;
    const int mBase  = blockIdx.y * 64 + waveM * 32;
    const int nBase  = blockIdx.x * 256 + waveN * 64;

    const _Float16* Xh = (const _Float16*)xptr;
    const float*    Xf = (const float*)xptr;

    const size_t xRow0 = ((size_t)t * BATCH + mBase + ln) * DIM + (size_t)laneHi * 16;
    const size_t xRow1 = xRow0 + (size_t)16 * DIM;
    size_t bRow[4];
#pragma unroll
    for (int nt = 0; nt < 4; ++nt)
        bRow[nt] = (size_t)(nBase + nt * 16 + ln) * DIM + (size_t)laneHi * 16;

    v8f acc[8];
#pragma unroll
    for (int i = 0; i < 8; ++i)
#pragma unroll
        for (int r = 0; r < 8; ++r) acc[i][r] = 0.0f;

#pragma unroll 2
    for (int k0 = 0; k0 < DIM; k0 += 32) {
        // Issue every load of the k-chunk first (distinct destinations).
        v16h a0 = XF16 ? ld_a_h(Xh + xRow0 + k0) : ld_a_f(Xf + xRow0 + k0);
        v16h a1 = XF16 ? ld_a_h(Xh + xRow1 + k0) : ld_a_f(Xf + xRow1 + k0);
        v16h b[4];
#pragma unroll
        for (int nt = 0; nt < 4; ++nt) b[nt] = ld_b_h(Wh + bRow[nt] + k0);
        // Then the 8-WMMA burst (4 independent chains per A operand).
#pragma unroll
        for (int nt = 0; nt < 4; ++nt) {
            acc[nt] = __builtin_amdgcn_wmma_f32_16x16x32_f16(
                false, a0, false, b[nt], (short)0, acc[nt], false, false);
            acc[4 + nt] = __builtin_amdgcn_wmma_f32_16x16x32_f16(
                false, a1, false, b[nt], (short)0, acc[4 + nt], false, false);
        }
    }

    // C/D layout: element r of subtile (ms, nt) ->
    //   row mBase + ms*16 + r + laneHi*8, col nBase + nt*16 + ln
    // C is streaming (written once, read once by the scan): non-temporal.
    const size_t cBase = ((size_t)t * BATCH + mBase + laneHi * 8) * DIM;
#pragma unroll
    for (int ms = 0; ms < 2; ++ms)
#pragma unroll
        for (int nt = 0; nt < 4; ++nt) {
            const int col = nBase + nt * 16 + ln;
#pragma unroll
            for (int r = 0; r < 8; ++r)
                __builtin_nontemporal_store(
                    acc[ms * 4 + nt][r],
                    &C[cBase + (size_t)(ms * 16 + r) * DIM + col]);
        }
}

// ============================================================================
// Kernel B: LIF scan over t (elementwise, float4 per thread)
// 512*2048/4 = 262144 threads; each runs the 20-step recurrence.
// ============================================================================
__global__ __launch_bounds__(256)
void lif_scan(const float* __restrict__ C, float* __restrict__ out) {
    const size_t q = (size_t)blockIdx.x * blockDim.x + threadIdx.x;
    const size_t planeq = (size_t)BATCH * DIM / 4;
    if (q >= planeq) return;
    float V0 = 0.f, V1 = 0.f, V2 = 0.f, V3 = 0.f;
    float I0 = 0.f, I1 = 0.f, I2 = 0.f, I3 = 0.f;
#pragma unroll
    for (int t = 0; t < STEPS; ++t) {
        const v4f c =
            __builtin_nontemporal_load(&((const v4f*)C)[(size_t)t * planeq + q]);
        v4f s;
        float Vn;
        Vn = AM * V0 + DTM * I0; s[0] = (Vn >= VTH) ? 1.0f : 0.0f; V0 = Vn * (1.0f - s[0]); I0 = AS * I0 + c[0];
        Vn = AM * V1 + DTM * I1; s[1] = (Vn >= VTH) ? 1.0f : 0.0f; V1 = Vn * (1.0f - s[1]); I1 = AS * I1 + c[1];
        Vn = AM * V2 + DTM * I2; s[2] = (Vn >= VTH) ? 1.0f : 0.0f; V2 = Vn * (1.0f - s[2]); I2 = AS * I2 + c[2];
        Vn = AM * V3 + DTM * I3; s[3] = (Vn >= VTH) ? 1.0f : 0.0f; V3 = Vn * (1.0f - s[3]); I3 = AS * I3 + c[3];
        __builtin_nontemporal_store(s, &((v4f*)out)[(size_t)t * planeq + q]);
    }
}

// ============================================================================
// Fallback: fused LIF + WMMA GEMM (t-sequential) for small workspace.
// block = 256 threads (8 waves). Wave tile 16x64; block tile 64x128.
// ============================================================================
template <bool WF16>
__global__ __launch_bounds__(256)
void lif_wmma_fused(const float* __restrict__ Xf,
                    const void* __restrict__ wptr,
                    float* __restrict__ out) {
    const int tid    = threadIdx.x;
    const int lane   = tid & 31;
    const int ln     = lane & 15;
    const int laneHi = lane >> 4;
    const int waveId = tid >> 5;
    const int waveM  = waveId & 3;
    const int waveN  = waveId >> 2;
    const int mBase  = blockIdx.y * 64 + waveM * 16;
    const int nBase  = blockIdx.x * 128 + waveN * 64;

    const _Float16* Wh = (const _Float16*)wptr;
    const float*    Wf = (const float*)wptr;

    v8f Vv[4], Iv[4];
#pragma unroll
    for (int nt = 0; nt < 4; ++nt)
#pragma unroll
        for (int r = 0; r < 8; ++r) { Vv[nt][r] = 0.0f; Iv[nt][r] = 0.0f; }

    const size_t aRow = (size_t)(mBase + ln) * DIM + (size_t)laneHi * 16;
    size_t bRow[4];
#pragma unroll
    for (int nt = 0; nt < 4; ++nt)
        bRow[nt] = (size_t)(nBase + nt * 16 + ln) * DIM + (size_t)laneHi * 16;

    for (int t = 0; t < STEPS; ++t) {
        const size_t xBase = (size_t)t * BATCH * DIM + aRow;
        v8f acc[4];
#pragma unroll
        for (int nt = 0; nt < 4; ++nt)
#pragma unroll
            for (int r = 0; r < 8; ++r) acc[nt][r] = 0.0f;

#pragma unroll 2
        for (int k0 = 0; k0 < DIM; k0 += 32) {
            v16h a = ld_a_f(Xf + xBase + k0);
            v16h b[4];
#pragma unroll
            for (int nt = 0; nt < 4; ++nt)
                b[nt] = WF16 ? ld_b_h(Wh + bRow[nt] + k0)
                             : ld_b_f(Wf + bRow[nt] + k0);
#pragma unroll
            for (int nt = 0; nt < 4; ++nt)
                acc[nt] = __builtin_amdgcn_wmma_f32_16x16x32_f16(
                    false, a, false, b[nt], (short)0, acc[nt], false, false);
        }

        const size_t outBase = ((size_t)t * BATCH + (size_t)(mBase + laneHi * 8)) * DIM;
#pragma unroll
        for (int nt = 0; nt < 4; ++nt) {
            const int col = nBase + nt * 16 + ln;
#pragma unroll
            for (int r = 0; r < 8; ++r) {
                float Vn = AM * Vv[nt][r] + DTM * Iv[nt][r];
                float s  = (Vn >= VTH) ? 1.0f : 0.0f;
                __builtin_nontemporal_store(
                    s, &out[outBase + (size_t)r * DIM + col]);
                Vv[nt][r] = Vn * (1.0f - s);
                Iv[nt][r] = AS * Iv[nt][r] + acc[nt][r];
            }
        }
    }
}

// ---------------------------------------------------------------------------
extern "C" void kernel_launch(void* const* d_in, const int* in_sizes, int n_in,
                              void* d_out, int out_size, void* d_ws, size_t ws_size,
                              hipStream_t stream) {
    const float* x = (const float*)d_in[0];   // [20, 512, 2048] spikes (f32)
    const float* W = (const float*)d_in[1];   // [2048, 2048] f32
    float* out = (float*)d_out;               // [20, 512, 2048] f32

    const size_t nX = (size_t)STEPS * BATCH * DIM;   // 20,971,520
    const size_t nW = (size_t)DIM * DIM;             // 4,194,304
    const size_t bW = nW * sizeof(_Float16);         //  8 MB  (W f16)
    const size_t bX = nX * sizeof(_Float16);         // 40 MB  (x f16)
    const size_t bC = nX * sizeof(float);            // 80 MB  (C f32)

    const dim3 gGemm(DIM / 256, BATCH / 64, STEPS);  // (8, 8, 20)
    const dim3 gFused(DIM / 128, BATCH / 64);        // (16, 8)
    const dim3 blk(256);
    const unsigned gScan = (unsigned)(((size_t)BATCH * DIM / 4 + 255) / 256);
    const unsigned gCvtW = (unsigned)((nW / 4 + 255) / 256);
    const unsigned gCvtX = (unsigned)((nX / 4 + 255) / 256);

    if (ws_size >= bW + bX + bC) {
        // Fast path: W->f16, x->f16, t-parallel GEMM into C, then LIF scan.
        _Float16* Whf = (_Float16*)d_ws;
        _Float16* Xhf = (_Float16*)((char*)d_ws + bW);
        float*    C   = (float*)((char*)d_ws + bW + bX);
        cvt_f32_to_f16<<<gCvtW, blk, 0, stream>>>(W, Whf, nW / 4);
        cvt_f32_to_f16<<<gCvtX, blk, 0, stream>>>(x, Xhf, nX / 4);
        gemm_xt_wmma<true><<<gGemm, blk, 0, stream>>>((const void*)Xhf, Whf, C);
        lif_scan<<<gScan, blk, 0, stream>>>(C, out);
    } else if (ws_size >= bW + bC) {
        // W->f16, x converted on the fly in the GEMM.
        _Float16* Whf = (_Float16*)d_ws;
        float*    C   = (float*)((char*)d_ws + bW);
        cvt_f32_to_f16<<<gCvtW, blk, 0, stream>>>(W, Whf, nW / 4);
        gemm_xt_wmma<false><<<gGemm, blk, 0, stream>>>((const void*)x, Whf, C);
        lif_scan<<<gScan, blk, 0, stream>>>(C, out);
    } else if (ws_size >= bW) {
        // Fused t-sequential kernel with f16 W.
        _Float16* Whf = (_Float16*)d_ws;
        cvt_f32_to_f16<<<gCvtW, blk, 0, stream>>>(W, Whf, nW / 4);
        lif_wmma_fused<true><<<gFused, blk, 0, stream>>>(x, (const void*)Whf, out);
    } else {
        // No workspace at all: everything on the fly.
        lif_wmma_fused<false><<<gFused, blk, 0, stream>>>(x, (const void*)W, out);
    }
}
